// BertSelfAttention_37349035606449
// MI455X (gfx1250) — matrix operand order repaired and checked by reading, not compiled, not run on hardware
//
#include <hip/hip_runtime.h>

// ---------------------------------------------------------------------------
// BERT self-attention (relative_key_query) for MI455X / gfx1250, wave32 WMMA.
// B=4, L=1024, D=1024, H=16, DH=64, P=1024 (dist_emb rows = 2047).
//
// Pipeline:
//   k0: elementwise f32->bf16 for X and dist_emb; W -> bf16 transposed
//   k1: fused QKV GEMM: both tiles staged by TDM (double-buffered, one
//       barrier per k-step), bf16 WMMA. Q,K stored [b,h,l,d]; V [b,h,d,l].
//   k2: flash attention: S = QK^T + gather(QE) + gather(KE); QE/KE are
//       per-tile WMMA GEMMs against a TDM-staged window of dist_emb rows.
//       K/V/E tiles double-buffered via TDM with D# padding for LDS strides.
// ---------------------------------------------------------------------------

typedef __attribute__((ext_vector_type(16))) __bf16 v16bf;
typedef __attribute__((ext_vector_type(8)))  float  v8f;
typedef __attribute__((ext_vector_type(4)))  unsigned int v4u;
typedef __attribute__((ext_vector_type(8)))  int v8i;
typedef __attribute__((ext_vector_type(4)))  int v4i;

union FragBF { uint4 q[2]; v16bf v; };

__device__ __forceinline__ unsigned short f2bf(float f) {
  union { __bf16 b; unsigned short u; } c;
  c.b = (__bf16)f;           // hardware cvt on gfx1250 (RNE)
  return c.u;
}

// Build a 16x(bf16) fragment from two 16B chunks (LDS or global).
__device__ __forceinline__ v16bf frag_ld(const unsigned short* p0,
                                         const unsigned short* p1) {
  FragBF f;
  f.q[0] = *(const uint4*)p0;
  f.q[1] = *(const uint4*)p1;
  return f.v;
}

#define WMMA_BF16(A, B, C) \
  __builtin_amdgcn_wmma_f32_16x16x32_bf16(false, (A), false, (B), (short)0, (C), false, false)

#define LOG2E 1.44269504088896340736f

// ---------------------------------------------------------------------------
// TDM: 2D tile load global -> LDS with optional LDS row padding.
// D# packing per CDNA5 ISA:
//  group0: [1:0]=count=1, [63:32]=lds_addr, [120:64]=global_addr, [127:126]=2
//  group1: [17:16]=data_size(1=2B), [20]=pad_enable, [24:22]=pad_interval,
//          [31:25]=pad_amount, [79:48]=tensor_dim0, [111:80]=tensor_dim1,
//          [127:112]=tile_dim0, [143:128]=tile_dim1, [207:160]=dim0_stride
// Dims/strides in data_size units. Groups 2/3 zero (2D tensor).
// ---------------------------------------------------------------------------
__device__ __forceinline__ void tdm_load_2d(unsigned lds_addr, const void* gp,
                                            unsigned tensor_d0, unsigned tensor_d1,
                                            unsigned tile_d0, unsigned tile_d1,
                                            unsigned stride0,
                                            unsigned pad_interval, unsigned pad_amount) {
  unsigned long long ga = (unsigned long long)(size_t)gp;
  v4u g0;
  g0[0] = 1u;                                              // count=1, user D#
  g0[1] = lds_addr;                                        // LDS byte address
  g0[2] = (unsigned)ga;                                    // global_addr[31:0]
  g0[3] = (unsigned)((ga >> 32) & 0x1FFFFFFull) | 0x80000000u;  // [56:32], type=2
  v8i g1;
  g1[0] = (int)((1u << 16) |                               // data_size = 2 bytes
                (1u << 20) |                               // pad_enable
                (pad_interval << 22) | (pad_amount << 25));
  g1[1] = (int)((tensor_d0 & 0xFFFFu) << 16);              // dim0[15:0]
  g1[2] = (int)(((tensor_d0 >> 16) & 0xFFFFu) | ((tensor_d1 & 0xFFFFu) << 16));
  g1[3] = (int)(((tensor_d1 >> 16) & 0xFFFFu) | ((tile_d0 & 0xFFFFu) << 16));
  g1[4] = (int)(tile_d1 & 0xFFFFu);                        // tile_dim2 = 0
  g1[5] = (int)stride0;                                    // dim0_stride[31:0]
  g1[6] = 0;
  g1[7] = 0;
  v4i z4 = {0, 0, 0, 0};
#if defined(__clang_major__) && __clang_major__ >= 23
  v8i z8 = {0, 0, 0, 0, 0, 0, 0, 0};
  __builtin_amdgcn_tensor_load_to_lds(g0, g1, z4, z4, z8, 0);
#else
  __builtin_amdgcn_tensor_load_to_lds(g0, g1, z4, z4, 0);
#endif
}

// ---------------------------------------------------------------------------
// Kernel 0a: elementwise f32 -> bf16 (used for X and dist_emb)
// ---------------------------------------------------------------------------
__global__ void cvt_kernel(const float* __restrict__ in,
                           unsigned short* __restrict__ out, int n) {
  int i = blockIdx.x * blockDim.x + threadIdx.x;
  if (i < n) out[i] = f2bf(in[i]);
}

// ---------------------------------------------------------------------------
// Kernel 0b: W[k][n] f32 -> Wt[n][k] bf16 (tiled transpose through LDS)
// ---------------------------------------------------------------------------
__global__ __launch_bounds__(256) void wt_kernel(
    const float* __restrict__ Wq, const float* __restrict__ Wk,
    const float* __restrict__ Wv, unsigned short* __restrict__ Wtq,
    unsigned short* __restrict__ Wtk, unsigned short* __restrict__ Wtv) {
  const int z = blockIdx.z;
  const float* src = (z == 0) ? Wq : (z == 1) ? Wk : Wv;
  unsigned short* dst = (z == 0) ? Wtq : (z == 1) ? Wtk : Wtv;
  __shared__ float t[32][33];
  const int k0 = blockIdx.x * 32, n0 = blockIdx.y * 32;
  const int lx = threadIdx.x, ly = threadIdx.y;
#pragma unroll
  for (int s = 0; s < 32; s += 8)
    t[ly + s][lx] = src[(size_t)(k0 + ly + s) * 1024 + n0 + lx];
  __syncthreads();
#pragma unroll
  for (int s = 0; s < 32; s += 8)
    dst[(size_t)(n0 + ly + s) * 1024 + k0 + lx] = f2bf(t[lx][ly + s]);
}

// ---------------------------------------------------------------------------
// Kernel 1: fused QKV projection GEMM, fully TDM-fed.
// Xbf[4096,1024] bf16 @ Wt[n][k] bf16 (+bias). blockIdx.z selects Q/K/V.
// 128x128 tile, 8 waves each 32x64; double-buffered LDS, 1 barrier/k-step.
// ---------------------------------------------------------------------------
__global__ __launch_bounds__(256) void qkv_kernel(
    const unsigned short* __restrict__ Xbf,
    const unsigned short* __restrict__ Wtq, const unsigned short* __restrict__ Wtk,
    const unsigned short* __restrict__ Wtv,
    const float* __restrict__ bq, const float* __restrict__ bk,
    const float* __restrict__ bv,
    unsigned short* __restrict__ Qb, unsigned short* __restrict__ Kb,
    unsigned short* __restrict__ Vt) {
  const int z = blockIdx.z;
  const unsigned short* Wt = (z == 0) ? Wtq : (z == 1) ? Wtk : Wtv;
  const float* bias = (z == 0) ? bq : (z == 1) ? bk : bv;
  unsigned short* OutQK = (z == 0) ? Qb : Kb;

  __shared__ unsigned short As[2][128 * 40];  // [m][k] bf16, TDM-padded 32->40
  __shared__ unsigned short Bs[2][128 * 40];  // [n][k] bf16, TDM-padded 32->40

  const int tid  = threadIdx.x;
  const int lane = tid & 31;
  const int w    = tid >> 5;
  const int wm   = w & 3;
  const int wn   = w >> 2;
  const int tileM = blockIdx.y * 128;
  const int tileN = blockIdx.x * 128;

  const int al  = lane & 15;
  const int ac  = (lane & 16) ? 8 : 0;
  const int bc  = (lane & 16) ? 16 : 0;
  const int hi8 = (lane & 16) ? 8 : 0;

  const unsigned ldsA[2] = {(unsigned)(size_t)(const void*)&As[0][0],
                            (unsigned)(size_t)(const void*)&As[1][0]};
  const unsigned ldsB[2] = {(unsigned)(size_t)(const void*)&Bs[0][0],
                            (unsigned)(size_t)(const void*)&Bs[1][0]};

  v8f acc[2][4];
#pragma unroll
  for (int mf = 0; mf < 2; mf++)
#pragma unroll
    for (int nf = 0; nf < 4; nf++) acc[mf][nf] = (v8f){};

  // Prologue: DMA first k-tiles into buffer 0.
  if (w == 0) {
    tdm_load_2d(ldsA[0], Xbf + (size_t)tileM * 1024,
                1024, 4096 - tileM, 32, 128, 1024, /*16 DW*/3, /*4 DW*/3);
    tdm_load_2d(ldsB[0], Wt + (size_t)tileN * 1024,
                1024, 1024 - tileN, 32, 128, 1024, 3, 3);
  }

  for (int k0 = 0; k0 < 1024; k0 += 32) {
    const int cur = (k0 >> 5) & 1;
    if (w == 0) __builtin_amdgcn_s_wait_tensorcnt(0);
    __syncthreads();  // current buffer ready for everyone; prior reads done
    if (w == 0 && k0 + 32 < 1024) {
      const int kn = k0 + 32;
      tdm_load_2d(ldsA[1 - cur], Xbf + (size_t)tileM * 1024 + kn,
                  1024 - kn, 4096 - tileM, 32, 128, 1024, 3, 3);
      tdm_load_2d(ldsB[1 - cur], Wt + (size_t)tileN * 1024 + kn,
                  1024 - kn, 1024 - tileN, 32, 128, 1024, 3, 3);
    }

    const unsigned short* Ac = As[cur];
    const unsigned short* Bc = Bs[cur];
    v16bf af[2], bfv[4];
#pragma unroll
    for (int mf = 0; mf < 2; mf++) {
      const unsigned short* pr = &Ac[(wm * 32 + mf * 16 + al) * 40];
      af[mf] = frag_ld(pr + ac, pr + ac + 16);
    }
#pragma unroll
    for (int nf = 0; nf < 4; nf++) {
      const unsigned short* pr = &Bc[(wn * 64 + nf * 16 + al) * 40];
      bfv[nf] = frag_ld(pr + ac, pr + ac + 16);
    }
#pragma unroll
    for (int mf = 0; mf < 2; mf++)
#pragma unroll
      for (int nf = 0; nf < 4; nf++)
        acc[mf][nf] = WMMA_BF16(af[mf], bfv[nf], acc[mf][nf]);
  }

  // Epilogue: + bias, bf16, scatter to head-major layouts.
#pragma unroll
  for (int mf = 0; mf < 2; mf++)
#pragma unroll
    for (int nf = 0; nf < 4; nf++)
#pragma unroll
      for (int i = 0; i < 8; i++) {
        int row = tileM + wm * 32 + mf * 16 + hi8 + i;   // b*1024 + l
        int col = tileN + wn * 64 + nf * 16 + al;        // h*64 + d
        float v = acc[mf][nf][i] + bias[col];
        unsigned short bvv = f2bf(v);
        int bb = row >> 10, l = row & 1023, hh = col >> 6, d = col & 63;
        if (z < 2)
          OutQK[((size_t)(bb * 16 + hh) * 1024 + l) * 64 + d] = bvv;
        else
          Vt[((size_t)(bb * 16 + hh) * 64 + d) * 1024 + l] = bvv;
      }
}

// ---------------------------------------------------------------------------
// Kernel 2: flash attention with relative_key_query bias.
// Grid (L/128, H, B); 8 waves; wave w owns rows l0+16w..+15. Column blocks
// of 32. Per block, TDM stages K (32x64), V^T (64x32) and a 160-row window
// of dist_emb into double-buffered padded LDS tiles, then WMMA: KE (whole
// WG), QE (per wave), S, and P.V with online softmax.
// ---------------------------------------------------------------------------
__global__ __launch_bounds__(256) void attn_kernel(
    const unsigned short* __restrict__ Qb, const unsigned short* __restrict__ Kb,
    const unsigned short* __restrict__ Vt, const unsigned short* __restrict__ Ebf,
    const float* __restrict__ mask, float* __restrict__ out) {
  __shared__ unsigned short Ks[2][32 * 72];    // [r][d], TDM-padded 64->72
  __shared__ unsigned short Vs[2][64 * 40];    // [d][r], TDM-padded 32->40
  __shared__ unsigned short Es[2][160 * 72];   // dist_emb window [j][d]
  __shared__ float KEs[32 * 164];              // [r][j], j in [0,160)
  __shared__ float QEs[8 * 16 * 52];           // per-wave [l][jw], jw in [0,48)
  __shared__ unsigned short Ps[8 * 16 * 40];   // per-wave P tile bf16
  __shared__ float maskS[32];

  const int tid = threadIdx.x, lane = tid & 31, w = tid >> 5;
  const int h = blockIdx.y, b = blockIdx.z;
  const int l0 = blockIdx.x * 128;
  const size_t baseQK = (size_t)(b * 16 + h) * 1024 * 64;
  const size_t baseV  = (size_t)(b * 16 + h) * 64 * 1024;

  const int al  = lane & 15;
  const int hi8 = (lane & 16) ? 8 : 0;
  const int ac  = (lane & 16) ? 8 : 0;
  const int bc  = (lane & 16) ? 16 : 0;

  const unsigned ldsKs[2] = {(unsigned)(size_t)(const void*)&Ks[0][0],
                             (unsigned)(size_t)(const void*)&Ks[1][0]};
  const unsigned ldsVs[2] = {(unsigned)(size_t)(const void*)&Vs[0][0],
                             (unsigned)(size_t)(const void*)&Vs[1][0]};
  const unsigned ldsEs[2] = {(unsigned)(size_t)(const void*)&Es[0][0],
                             (unsigned)(size_t)(const void*)&Es[1][0]};

  // Q fragments for this wave's 16 rows (2 k-steps over DH=64), loaded once.
  v16bf qa[2];
  {
    const unsigned short* qp = Qb + baseQK + (size_t)(l0 + w * 16 + al) * 64;
    qa[0] = frag_ld(qp + ac, qp + ac + 16);
    qa[1] = frag_ld(qp + 32 + ac, qp + 32 + ac + 16);
  }

  v8f ctx[4];
#pragma unroll
  for (int nf = 0; nf < 4; nf++) ctx[nf] = (v8f){};
  float mrow[8], lrow[8];
#pragma unroll
  for (int i = 0; i < 8; i++) { mrow[i] = -__builtin_inff(); lrow[i] = 0.f; }

  float* QEw = &QEs[w * 16 * 52];
  unsigned short* Pw = &Ps[w * 16 * 40];

  // Prologue: DMA first column block into buffer 0.
  if (w == 0) {
    const int jb0 = l0 + 992;
    tdm_load_2d(ldsKs[0], Kb + baseQK, 64, 1024, 64, 32, 64, /*32 DW*/4, /*4 DW*/3);
    tdm_load_2d(ldsVs[0], Vt + baseV, 1024, 64, 32, 64, 1024, 3, 3);
    tdm_load_2d(ldsEs[0], Ebf + (size_t)jb0 * 64, 64, 2047u - (unsigned)jb0,
                64, 160, 64, 4, 3);
  }

  for (int r0 = 0; r0 < 1024; r0 += 32) {
    const int cur = (r0 >> 5) & 1;
    if (w == 0) __builtin_amdgcn_s_wait_tensorcnt(0);
    __syncthreads();  // (1) tiles ready; prior iteration fully consumed

    if (w == 0 && r0 + 32 < 1024) {
      const int rn = r0 + 32;
      const int jbn = l0 - rn + 992;
      tdm_load_2d(ldsKs[1 - cur], Kb + baseQK + (size_t)rn * 64,
                  64, 1024 - rn, 64, 32, 64, 4, 3);
      tdm_load_2d(ldsVs[1 - cur], Vt + baseV + rn,
                  1024 - (unsigned)rn, 64, 32, 64, 1024, 3, 3);
      tdm_load_2d(ldsEs[1 - cur], Ebf + (size_t)jbn * 64,
                  64, 2047u - (unsigned)jbn, 64, 160, 64, 4, 3);
    }
    if (tid < 32) maskS[tid] = mask[b * 1024 + r0 + tid];

    const unsigned short* Kc = Ks[cur];
    const unsigned short* Vc = Vs[cur];
    const unsigned short* Ec = Es[cur];

    // KE[r][j] = k[r0+r] . E[jbase+j]  — 20 frag positions over 8 waves.
    for (int p = w; p < 20; p += 8) {
      int mf = p & 1, nf = p >> 1;
      v8f a = (v8f){};
#pragma unroll
      for (int ks = 0; ks < 2; ks++) {
        const unsigned short* ap = &Kc[(mf * 16 + al) * 72 + ks * 32];
        v16bf af = frag_ld(ap + ac, ap + ac + 16);
        const unsigned short* bp = &Ec[(nf * 16 + al) * 72 + ks * 32];
        v16bf bv = frag_ld(bp + ac, bp + ac + 16);
        a = WMMA_BF16(af, bv, a);
      }
#pragma unroll
      for (int i = 0; i < 8; i++)
        KEs[(mf * 16 + hi8 + i) * 164 + nf * 16 + al] = a[i];
    }

    // QE[l][jw] = q[l] . E[jbase + 16w + jw]  (per-wave 48-wide window)
#pragma unroll
    for (int nf = 0; nf < 3; nf++) {
      v8f a = (v8f){};
#pragma unroll
      for (int ks = 0; ks < 2; ks++) {
        const unsigned short* bp = &Ec[(w * 16 + nf * 16 + al) * 72 + ks * 32];
        v16bf bv = frag_ld(bp + ac, bp + ac + 16);
        a = WMMA_BF16(qa[ks], bv, a);
      }
#pragma unroll
      for (int i = 0; i < 8; i++)
        QEw[(hi8 + i) * 52 + nf * 16 + al] = a[i];
    }

    // S = Q K^T (16x32 tile per wave)
    v8f s[2]; s[0] = (v8f){}; s[1] = (v8f){};
#pragma unroll
    for (int ks = 0; ks < 2; ks++)
#pragma unroll
      for (int fc = 0; fc < 2; fc++) {
        const unsigned short* bp = &Kc[(fc * 16 + al) * 72 + ks * 32];
        v16bf bv = frag_ld(bp + ac, bp + ac + 16);
        s[fc] = WMMA_BF16(qa[ks], bv, s[fc]);
      }

    __syncthreads();  // (2) KEs/maskS visible to all waves

    // Assemble scores with anti-diagonal gathers, then online softmax.
    float t0[8], t1[8], p0[8], p1[8], corr[8];
#pragma unroll
    for (int i = 0; i < 8; i++) {
      int rl = hi8 + i;
      { int rc = al;      int jw = rl - rc + 31; int jg = w * 16 + rl - rc + 31;
        t0[i] = (s[0][i] + QEw[rl * 52 + jw] + KEs[rc * 164 + jg]) * 0.125f + maskS[rc]; }
      { int rc = 16 + al; int jw = rl - rc + 31; int jg = w * 16 + rl - rc + 31;
        t1[i] = (s[1][i] + QEw[rl * 52 + jw] + KEs[rc * 164 + jg]) * 0.125f + maskS[rc]; }
    }
#pragma unroll
    for (int i = 0; i < 8; i++) {
      float mx = fmaxf(t0[i], t1[i]);
      mx = fmaxf(mx, __shfl_xor(mx, 1));
      mx = fmaxf(mx, __shfl_xor(mx, 2));
      mx = fmaxf(mx, __shfl_xor(mx, 4));
      mx = fmaxf(mx, __shfl_xor(mx, 8));
      float mn = fmaxf(mrow[i], mx);
      corr[i] = exp2f((mrow[i] - mn) * LOG2E);
      mrow[i] = mn;
      p0[i] = exp2f((t0[i] - mn) * LOG2E);
      p1[i] = exp2f((t1[i] - mn) * LOG2E);
      float ps = p0[i] + p1[i];
      ps += __shfl_xor(ps, 1);
      ps += __shfl_xor(ps, 2);
      ps += __shfl_xor(ps, 4);
      ps += __shfl_xor(ps, 8);
      lrow[i] = lrow[i] * corr[i] + ps;
    }
#pragma unroll
    for (int nf = 0; nf < 4; nf++)
#pragma unroll
      for (int i = 0; i < 8; i++) ctx[nf][i] *= corr[i];

    // P (C-layout) -> LDS -> A-layout fragment, then ctx += P . V
#pragma unroll
    for (int i = 0; i < 8; i++) {
      Pw[(hi8 + i) * 40 + al]      = f2bf(p0[i]);
      Pw[(hi8 + i) * 40 + 16 + al] = f2bf(p1[i]);
    }
    const unsigned short* pp = &Pw[al * 40];
    v16bf pa = frag_ld(pp + ac, pp + ac + 16);
#pragma unroll
    for (int nf = 0; nf < 4; nf++) {
      const unsigned short* vp = &Vc[(nf * 16 + al) * 40];
      v16bf bv = frag_ld(vp + ac, vp + ac + 16);
      ctx[nf] = WMMA_BF16(pa, bv, ctx[nf]);
    }
  }

  // out[b][l][h*64+d] = ctx / l
#pragma unroll
  for (int nf = 0; nf < 4; nf++)
#pragma unroll
    for (int i = 0; i < 8; i++) {
      int lg = l0 + w * 16 + hi8 + i;
      int d = nf * 16 + al;
      out[(size_t)(b * 1024 + lg) * 1024 + h * 64 + d] = ctx[nf][i] / lrow[i];
    }
}

// ---------------------------------------------------------------------------
extern "C" void kernel_launch(void* const* d_in, const int* in_sizes, int n_in,
                              void* d_out, int out_size, void* d_ws, size_t ws_size,
                              hipStream_t stream) {
  (void)in_sizes; (void)n_in; (void)out_size; (void)ws_size;
  const float* X    = (const float*)d_in[0];
  const float* mask = (const float*)d_in[1];
  const float* Wq   = (const float*)d_in[2];
  const float* bq   = (const float*)d_in[3];
  const float* Wk   = (const float*)d_in[4];
  const float* bk   = (const float*)d_in[5];
  const float* Wv   = (const float*)d_in[6];
  const float* bv   = (const float*)d_in[7];
  const float* dist = (const float*)d_in[8];
  float* out = (float*)d_out;

  char* ws = (char*)d_ws;
  unsigned short* Qb  = (unsigned short*)(ws);                        // 8 MiB
  unsigned short* Kb  = (unsigned short*)(ws + (8u << 20));           // 8 MiB
  unsigned short* Vt  = (unsigned short*)(ws + (16u << 20));          // 8 MiB
  unsigned short* Ebf = (unsigned short*)(ws + (24u << 20));          // 256 KiB
  unsigned short* Xbf = (unsigned short*)(ws + (25u << 20));          // 8 MiB
  unsigned short* Wtq = (unsigned short*)(ws + (33u << 20));          // 2 MiB
  unsigned short* Wtk = (unsigned short*)(ws + (35u << 20));          // 2 MiB
  unsigned short* Wtv = (unsigned short*)(ws + (37u << 20));          // 2 MiB

  cvt_kernel<<<512, 256, 0, stream>>>(dist, Ebf, 2047 * 64);
  cvt_kernel<<<16384, 256, 0, stream>>>(X, Xbf, 4096 * 1024);
  wt_kernel<<<dim3(32, 32, 3), dim3(32, 8), 0, stream>>>(Wq, Wk, Wv, Wtq, Wtk, Wtv);
  qkv_kernel<<<dim3(8, 32, 3), 256, 0, stream>>>(Xbf, Wtq, Wtk, Wtv, bq, bk, bv,
                                                 Qb, Kb, Vt);
  attn_kernel<<<dim3(8, 16, 4), 256, 0, stream>>>(Qb, Kb, Vt, Ebf, mask, out);
}
